// biLSTM_dynamic_maxpool_82394652606544
// MI455X (gfx1250) — matrix operand order repaired and checked by reading, not compiled, not run on hardware
//
#include <hip/hip_runtime.h>
#include <hip/hip_bf16.h>
#include <float.h>

typedef __bf16 bf16_t;
typedef __attribute__((ext_vector_type(16))) __bf16 v16bf;
typedef __attribute__((ext_vector_type(8)))  __bf16 v8bf;
typedef __attribute__((ext_vector_type(8)))  float  v8f;
typedef __attribute__((ext_vector_type(4)))  float  v4f;

#define Bq   64
#define Tq   2048
#define Eq   128
#define H2q  128
#define G4   512   // 4*H2
#define Sq   8

// Dynamic-LDS layout (bytes)
static constexpr unsigned LDS_WIH   = 0;                    // 512*128 bf16 = 131072
static constexpr unsigned LDS_WHH   = 131072;               // 512*128 bf16 = 131072
static constexpr unsigned LDS_G     = 262144;               // 16*512 f32   = 32768
static constexpr unsigned LDS_H     = 262144 + 32768;       // 16*128 bf16  = 4096
static constexpr unsigned LDS_X     = LDS_H + 4096;         // 16*128 bf16  = 4096
static constexpr unsigned LDS_TOTAL = LDS_X + 4096;         // 303104 < 320K

__device__ __forceinline__ unsigned umin_(unsigned a, unsigned b) { return a < b ? a : b; }

// sigmoid / tanh built on the hardware v_exp_f32; both saturate cleanly at +-inf
__device__ __forceinline__ float sigmoidf_(float x) {
    return __fdividef(1.0f, 1.0f + __expf(-x));
}
__device__ __forceinline__ float tanhf_(float x) {
    return 1.0f - __fdividef(2.0f, __expf(2.0f * x) + 1.0f);
}

// assemble a 16-element bf16 WMMA fragment from two 16-byte LDS chunks
__device__ __forceinline__ v16bf load_frag16(const bf16_t* p0, const bf16_t* p1) {
    union { v16bf v; v8bf h2[2]; } u;
    u.h2[0] = *(const v8bf*)p0;
    u.h2[1] = *(const v8bf*)p1;
    return u.v;
}

__global__ void __launch_bounds__(256, 1)
bilstm_pool_kernel(const int* __restrict__ sentence,
                   const int* __restrict__ seq_lengths,
                   const float* __restrict__ emb,
                   const float* __restrict__ W_ih_f, const float* __restrict__ W_hh_f,
                   const float* __restrict__ b_f,
                   const float* __restrict__ W_ih_b, const float* __restrict__ W_hh_b,
                   const float* __restrict__ b_b,
                   float* __restrict__ pooled)
{
    extern __shared__ char smem[];
    bf16_t* lds_wih = (bf16_t*)(smem + LDS_WIH);   // [512][128] row-major (n,k)
    bf16_t* lds_whh = (bf16_t*)(smem + LDS_WHH);   // [512][128]
    float*  lds_g   = (float*) (smem + LDS_G);     // [16][512] gate pre-activations
    bf16_t* lds_h   = (bf16_t*)(smem + LDS_H);     // [16][128] hidden state (bf16)
    bf16_t* lds_x   = (bf16_t*)(smem + LDS_X);     // [16][128] current x tile (bf16)

    const int tid = threadIdx.x;
    const int dir = blockIdx.x >> 2;               // 0 = forward, 1 = backward
    const int b0  = (blockIdx.x & 3) * 16;         // batch-row slice base

    const float* Wih = dir ? W_ih_b : W_ih_f;
    const float* Whh = dir ? W_hh_b : W_hh_f;
    const float* bv  = dir ? b_b    : b_f;

    // ---- stage weights to LDS as bf16 (row-major [n=gate col][k]) ----
    for (int i = tid; i < G4 * H2q; i += 256) {
        lds_wih[i] = (bf16_t)Wih[i];
        lds_whh[i] = (bf16_t)Whh[i];
    }
    for (int i = tid; i < 16 * H2q; i += 256) lds_h[i] = (bf16_t)0.0f;

    // ---- element-wise thread mapping: r = local batch row, c0 = hidden col base ----
    const int r  = tid >> 4;            // 0..15
    const int c0 = (tid & 15) * 8;      // 0,8,...,120
    const unsigned len = (unsigned)seq_lengths[b0 + r];
    const unsigned w   = (len + 7u) >> 3;           // per-sample pool window

    float creg[8];                                   // cell state
    float cur[8];                                    // running max of current segment
    float mx[Sq][8];                                 // per-segment maxes (flushed rarely)
    unsigned curseg = dir ? 7u : 0u;                 // bwd starts in segment 7
    #pragma unroll
    for (int j = 0; j < 8; ++j) { creg[j] = 0.0f; cur[j] = -FLT_MAX; }
    for (int s = 0; s < Sq; ++s)
        #pragma unroll
        for (int j = 0; j < 8; ++j) mx[s][j] = -FLT_MAX;

    // ---- WMMA wave mapping ----
    const int lane  = tid & 31;
    const int wv    = tid >> 5;         // 0..7
    const int half  = lane >> 4;        // lane group
    const int m     = lane & 15;        // row (A) / col (B,C)
    const int nwave = wv * 64;          // this wave's 64 gate columns

    // per-lane bias for the wave's 4 tiles (loop invariant)
    const float bias0 = bv[nwave      + m];
    const float bias1 = bv[nwave + 16 + m];
    const float bias2 = bv[nwave + 32 + m];
    const float bias3 = bv[nwave + 48 + m];

    // loop-invariant fragment base pointers
    const bf16_t* xbase = lds_x + m * Eq;
    const bf16_t* hbase = lds_h + m * H2q;
    const bf16_t* wihb  = lds_wih + (nwave + m) * H2q;   // + i4*16*H2q per tile
    const bf16_t* whhb  = lds_whh + (nwave + m) * H2q;

    __syncthreads();

    // ---- software-pipelined embedding prefetch for step 0 ----
    v4f px0, px1;
    {
        const int t0  = dir ? (Tq - 1) : 0;
        const int tok = sentence[(b0 + r) * Tq + t0];
        const float* src = emb + (size_t)tok * Eq + c0;
        px0 = *(const v4f*)src;
        px1 = *(const v4f*)(src + 4);
    }

    for (int tt = 0; tt < Tq; ++tt) {
        const int t = dir ? (Tq - 1 - tt) : tt;

        // ---- commit prefetched x tile (row r, cols c0..c0+7) into LDS as bf16 ----
        {
            v8bf xv;
            #pragma unroll
            for (int j = 0; j < 4; ++j) { xv[j] = (bf16_t)px0[j]; xv[4 + j] = (bf16_t)px1[j]; }
            *(v8bf*)(lds_x + r * Eq + c0) = xv;
        }
        __syncthreads();

        // ---- prefetch next step's embedding; latency hidden under the WMMA phase ----
        {
            const int ttn = (tt + 1 < Tq) ? (tt + 1) : tt;
            const int tn  = dir ? (Tq - 1 - ttn) : ttn;
            const int tokn = sentence[(b0 + r) * Tq + tn];
            const float* srcn = emb + (size_t)tokn * Eq + c0;
            px0 = *(const v4f*)srcn;
            px1 = *(const v4f*)(srcn + 4);
        }

        // ---- WMMA: g = x @ Wih^T + h @ Whh^T + b ----
        // K-outer / tile-inner: one shared A fragment feeds 4 independent
        // accumulator chains (hides WMMA D->C RAW hazard latency).
        v8f acc0, acc1, acc2, acc3;
        #pragma unroll
        for (int rr = 0; rr < 8; ++rr) {
            acc0[rr] = bias0; acc1[rr] = bias1; acc2[rr] = bias2; acc3[rr] = bias3;
        }

        #pragma unroll
        for (int ks = 0; ks < 4; ++ks) {             // input projection
            const bf16_t* ap = xbase + ks * 32 + half * 16;
            const v16bf a = load_frag16(ap, ap + 8);
            const bf16_t* bp = wihb + ks * 32 + half * 16;
            const v16bf bf0 = load_frag16(bp,              bp + 8);
            const v16bf bf1 = load_frag16(bp + 16 * H2q,   bp + 16 * H2q + 8);
            const v16bf bf2 = load_frag16(bp + 32 * H2q,   bp + 32 * H2q + 8);
            const v16bf bf3 = load_frag16(bp + 48 * H2q,   bp + 48 * H2q + 8);
            acc0 = __builtin_amdgcn_wmma_f32_16x16x32_bf16(false, a, false, bf0, (short)0, acc0, false, false);
            acc1 = __builtin_amdgcn_wmma_f32_16x16x32_bf16(false, a, false, bf1, (short)0, acc1, false, false);
            acc2 = __builtin_amdgcn_wmma_f32_16x16x32_bf16(false, a, false, bf2, (short)0, acc2, false, false);
            acc3 = __builtin_amdgcn_wmma_f32_16x16x32_bf16(false, a, false, bf3, (short)0, acc3, false, false);
        }
        #pragma unroll
        for (int ks = 0; ks < 4; ++ks) {             // recurrent projection
            const bf16_t* ap = hbase + ks * 32 + half * 16;
            const v16bf a = load_frag16(ap, ap + 8);
            const bf16_t* bp = whhb + ks * 32 + half * 16;
            const v16bf bf0 = load_frag16(bp,              bp + 8);
            const v16bf bf1 = load_frag16(bp + 16 * H2q,   bp + 16 * H2q + 8);
            const v16bf bf2 = load_frag16(bp + 32 * H2q,   bp + 32 * H2q + 8);
            const v16bf bf3 = load_frag16(bp + 48 * H2q,   bp + 48 * H2q + 8);
            acc0 = __builtin_amdgcn_wmma_f32_16x16x32_bf16(false, a, false, bf0, (short)0, acc0, false, false);
            acc1 = __builtin_amdgcn_wmma_f32_16x16x32_bf16(false, a, false, bf1, (short)0, acc1, false, false);
            acc2 = __builtin_amdgcn_wmma_f32_16x16x32_bf16(false, a, false, bf2, (short)0, acc2, false, false);
            acc3 = __builtin_amdgcn_wmma_f32_16x16x32_bf16(false, a, false, bf3, (short)0, acc3, false, false);
        }

        // C/D layout: element rr -> row rr + 8*half, col (tile base + m)
        #pragma unroll
        for (int rr = 0; rr < 8; ++rr) {
            float* go = lds_g + (rr + half * 8) * G4 + nwave + m;
            go[0]  = acc0[rr];
            go[16] = acc1[rr];
            go[32] = acc2[rr];
            go[48] = acc3[rr];
        }
        __syncthreads();

        // ---- LSTM cell + fused ragged max-pool (valid rows only) ----
        if ((unsigned)t < len) {
            const unsigned seg = umin_((unsigned)t / w, 7u);
            if (seg != curseg) {
                #pragma unroll
                for (int j = 0; j < 8; ++j) {
                    mx[curseg][j] = fmaxf(mx[curseg][j], cur[j]);
                    cur[j] = -FLT_MAX;
                }
                curseg = seg;
            }
            const float* gp = lds_g + r * G4 + c0;
            v4f gi0 = *(const v4f*)(gp);        v4f gi1 = *(const v4f*)(gp + 4);
            v4f gf0 = *(const v4f*)(gp + 128);  v4f gf1 = *(const v4f*)(gp + 132);
            v4f gg0 = *(const v4f*)(gp + 256);  v4f gg1 = *(const v4f*)(gp + 260);
            v4f go0 = *(const v4f*)(gp + 384);  v4f go1 = *(const v4f*)(gp + 388);
            v8bf hv;
            #pragma unroll
            for (int j = 0; j < 8; ++j) {
                const float gi = (j < 4) ? gi0[j & 3] : gi1[j & 3];
                const float gf = (j < 4) ? gf0[j & 3] : gf1[j & 3];
                const float gg = (j < 4) ? gg0[j & 3] : gg1[j & 3];
                const float go = (j < 4) ? go0[j & 3] : go1[j & 3];
                const float cn = sigmoidf_(gf) * creg[j] + sigmoidf_(gi) * tanhf_(gg);
                const float hn = sigmoidf_(go) * tanhf_(cn);
                creg[j] = cn;
                hv[j]   = (bf16_t)hn;
                cur[j]  = fmaxf(cur[j], hn);
            }
            *(v8bf*)(lds_h + r * H2q + c0) = hv;   // masked rows keep previous h
        }
        __syncthreads();
    }

    // final flush of the last segment
    #pragma unroll
    for (int j = 0; j < 8; ++j) mx[curseg][j] = fmaxf(mx[curseg][j], cur[j]);

    // pad-floor (segments overlapping the zero-padded tail include value 0) + store
    const unsigned lenDivW = len / w;
    const bool padded = (w * 8u > len);
    for (int s = 0; s < Sq; ++s) {
        const bool hp = padded && ((unsigned)s >= lenDivW);
        #pragma unroll
        for (int j = 0; j < 8; ++j) {
            float v = mx[s][j];
            if (hp) v = fmaxf(v, 0.0f);
            pooled[((size_t)(b0 + r) * Sq + s) * 256 + dir * 128 + c0 + j] = v;
        }
    }
}

// out[b,c] = sum_{h,s} pooled[b,s,h] * Wd[c, h*8+s] + bd[c]
__global__ void __launch_bounds__(256)
dense_kernel(const float* __restrict__ pooled,
             const float* __restrict__ Wd,
             const float* __restrict__ bd,
             float* __restrict__ out)
{
    __shared__ float red0[256];
    __shared__ float red1[256];
    const int b = blockIdx.x, tid = threadIdx.x;
    float a0 = 0.0f, a1 = 0.0f;
    for (int k = tid; k < 2048; k += 256) {          // k = h*8 + s
        const int hh = k >> 3, s = k & 7;
        const float v = pooled[((size_t)b * 8 + s) * 256 + hh];
        a0 += v * Wd[k];
        a1 += v * Wd[2048 + k];
    }
    red0[tid] = a0; red1[tid] = a1;
    __syncthreads();
    for (int off = 128; off > 0; off >>= 1) {
        if (tid < off) { red0[tid] += red0[tid + off]; red1[tid] += red1[tid + off]; }
        __syncthreads();
    }
    if (tid == 0) {
        out[b * 2 + 0] = red0[0] + bd[0];
        out[b * 2 + 1] = red1[0] + bd[1];
    }
}

extern "C" void kernel_launch(void* const* d_in, const int* in_sizes, int n_in,
                              void* d_out, int out_size, void* d_ws, size_t ws_size,
                              hipStream_t stream) {
    (void)in_sizes; (void)n_in; (void)out_size; (void)ws_size;
    const int*   sentence = (const int*)  d_in[0];
    const int*   lens     = (const int*)  d_in[1];
    const float* emb      = (const float*)d_in[2];
    const float* W_ih_f   = (const float*)d_in[3];
    const float* W_hh_f   = (const float*)d_in[4];
    const float* b_f      = (const float*)d_in[5];
    const float* W_ih_b   = (const float*)d_in[6];
    const float* W_hh_b   = (const float*)d_in[7];
    const float* b_b      = (const float*)d_in[8];
    const float* W_dense  = (const float*)d_in[9];
    const float* b_dense  = (const float*)d_in[10];

    float* pooled = (float*)d_ws;          // [64][8][256] f32 = 512 KB
    float* out    = (float*)d_out;         // [64][2] f32

    bilstm_pool_kernel<<<dim3(8), dim3(256), LDS_TOTAL, stream>>>(
        sentence, lens, emb, W_ih_f, W_hh_f, b_f, W_ih_b, W_hh_b, b_b, pooled);

    dense_kernel<<<dim3(64), dim3(256), 0, stream>>>(pooled, W_dense, b_dense, out);
}
